// Mamba_9689446220430
// MI455X (gfx1250) — matrix operand order repaired and checked by reading, not compiled, not run on hardware
//
#include <hip/hip_runtime.h>

// ---------------------------------------------------------------------------
// Bidirectional Mamba block for MI455X (gfx1250, wave32, WMMA + TDM).
// GEMMs in bf16 -> f32 via v_wmma_f32_16x16x32_bf16; tile staging via
// tensor_load_to_lds (Tensor Data Mover) with double-buffered LDS; scan in f32.
// ---------------------------------------------------------------------------

#define BATCH   4
#define SEQ     2048
#define DMODEL  1024
#define DINNER  2048
#define DSTATE  16
#define DTRANK  64
#define EDIM    96            // DTRANK + 2*DSTATE
#define BL      (BATCH*SEQ)   // 8192 rows for all GEMMs

typedef __attribute__((ext_vector_type(8)))  __bf16 bf16x8;
typedef __attribute__((ext_vector_type(16))) __bf16 bf16x16;
typedef __attribute__((ext_vector_type(8)))  float  v8f;
typedef __attribute__((ext_vector_type(4)))  unsigned int u32x4;
typedef __attribute__((ext_vector_type(8)))  int i32x8;
typedef __attribute__((ext_vector_type(4)))  int i32x4;

struct U16B { unsigned long long a, b; };   // 16-byte copy unit

#if defined(__gfx1250__) && __has_builtin(__builtin_amdgcn_tensor_load_to_lds)
#define USE_TDM 1
#else
#define USE_TDM 0
#endif

__device__ __forceinline__ unsigned short f32_to_bf16(float f) {
  unsigned int u = __float_as_uint(f);
  unsigned int r = u + 0x7FFFu + ((u >> 16) & 1u);   // round-to-nearest-even
  return (unsigned short)(r >> 16);
}
__device__ __forceinline__ float bf16_to_f32(unsigned short h) {
  return __uint_as_float(((unsigned int)h) << 16);
}
__device__ __forceinline__ float silu(float v) {
  return v / (1.0f + __expf(-v));
}

#if USE_TDM
// ---------------------------------------------------------------------------
// Issue one 2-D TDM tile load: tile_rows rows of tile_cols bf16 elements,
// global row stride `stride` (elements), into LDS at byte address lds_addr.
// avail_rows/avail_cols clip the tensor so out-of-range reads return zero.
// D# layout per CDNA5 ISA ch. 8 (group0 128b, group1 256b).
// ---------------------------------------------------------------------------
__device__ __forceinline__ void tdm_load_tile_2d(
    unsigned lds_addr, const unsigned short* gptr,
    unsigned tile_cols, unsigned tile_rows,
    unsigned long long stride, unsigned avail_cols, unsigned avail_rows) {
  unsigned long long ga = (unsigned long long)(uintptr_t)gptr;
  u32x4 g0 = (u32x4)0u;
  g0[0] = 1u;                                            // count=1 (user D#)
  g0[1] = lds_addr;                                      // lds_addr[31:0]
  g0[2] = (unsigned)(ga & 0xFFFFFFFFull);                // global_addr[31:0]
  g0[3] = (unsigned)((ga >> 32) & 0x1FFFFFFull)          // global_addr[56:32]
        | (2u << 30);                                    // type=2 ("image")
  unsigned long long sb = stride;                        // data_size units
  i32x8 g1 = (i32x8)0;
  g1[0] = (int)(1u << 16);                               // data_size=1 -> 2B
  g1[1] = (int)((avail_cols & 0xFFFFu) << 16);           // tensor_dim0[15:0]
  g1[2] = (int)((avail_cols >> 16) |
                ((avail_rows & 0xFFFFu) << 16));         // td0 hi | td1 lo
  g1[3] = (int)((avail_rows >> 16) |
                ((tile_cols & 0xFFFFu) << 16));          // td1 hi | tile_dim0
  g1[4] = (int)(tile_rows & 0xFFFFu);                    // tile_dim1 (dim2=0)
  g1[5] = (int)(sb & 0xFFFFFFFFull);                     // dim0_stride[31:0]
  g1[6] = (int)((sb >> 32) & 0xFFFFull);                 // dim0_stride[47:32]
  g1[7] = 0;
  i32x4 z4 = (i32x4)0;
#if __clang_major__ >= 23
  i32x8 z8 = (i32x8)0;
  __builtin_amdgcn_tensor_load_to_lds(g0, g1, z4, z4, z8, 0);
#else
  __builtin_amdgcn_tensor_load_to_lds(g0, g1, z4, z4, 0);
#endif
}
#endif  // USE_TDM

// ---------------------------------------------------------------------------
// casts
// ---------------------------------------------------------------------------
__global__ __launch_bounds__(256) void cast_f2b(const float* __restrict__ s,
                                                unsigned short* __restrict__ d,
                                                long n) {
  long i = (long)blockIdx.x * 256 + threadIdx.x;
  if (i < n) d[i] = f32_to_bf16(s[i]);
}

// gather a column sub-block (first `cols` of each row with stride sld) -> bf16
__global__ __launch_bounds__(256) void cast_cols(const float* __restrict__ s,
                                                 unsigned short* __restrict__ d,
                                                 long rows, int sld, int cols) {
  long i = (long)blockIdx.x * 256 + threadIdx.x;
  if (i >= rows * (long)cols) return;
  long r = i / cols;
  int  c = (int)(i - r * cols);
  d[i] = f32_to_bf16(s[r * sld + c]);
}

// ---------------------------------------------------------------------------
// Generic WMMA GEMM:  C(MxN) = A(MxK, bf16 row-major, lda) * Bt(NxK, bf16)^T
// EPI: 0 = none, 1 = softplus(x + bias[n])
// ST : 0 = row-major C[m*ldc+n], 1 = "bel": m=(b,l) -> C[(b*ldc+n)*SEQ+l]
// M multiple of 128, K multiple of 32; N guarded.
// Tile staging via TDM (double-buffered) when available.
// ---------------------------------------------------------------------------
template <int EPI, int ST>
__global__ __launch_bounds__(256) void wmma_gemm(
    const unsigned short* __restrict__ A, int lda,
    const unsigned short* __restrict__ Bt,
    float* __restrict__ C, long ldc,
    int M, int N, int K,
    const float* __restrict__ bias) {
  constexpr int BM = 128, BN = 64, BK = 32;
  constexpr int ATILE = BM * BK;           // ushort elements
  constexpr int TILE  = ATILE + BN * BK;   // one double-buffer slot
  __shared__ __align__(16) unsigned short SMEM[2 * TILE];

  const int tid  = threadIdx.x;
  const int lane = tid & 31;
  const int wave = tid >> 5;
  const int wm   = wave & 3;   // 4 waves along M (32 rows each)
  const int wn   = wave >> 2;  // 2 waves along N (32 cols each)
  const int bm   = blockIdx.y * BM;
  const int bn   = blockIdx.x * BN;
  const int half = lane >> 4;
  const int l16  = lane & 15;
  const int nk   = K / BK;

  v8f acc[2][2] = {};

#if USE_TDM
  const unsigned smem_base = (unsigned)(uintptr_t)&SMEM[0];
  const bool issuer = (wave == 0);
  // prologue: DMA tile 0 into buffer 0
  if (issuer) {
    tdm_load_tile_2d(smem_base, A + (long)bm * lda, BK, BM,
                     (unsigned long long)lda, (unsigned)(K), (unsigned)(M - bm));
    tdm_load_tile_2d(smem_base + 2u * ATILE, Bt + (long)bn * K, BK, BN,
                     (unsigned long long)K, (unsigned)(K),
                     (unsigned)(N > bn ? N - bn : 0));
  }
#endif

  for (int ki = 0; ki < nk; ++ki) {
    const int k0 = ki * BK;
#if USE_TDM
    const int p = ki & 1;
    unsigned short* As = &SMEM[p * TILE];
    unsigned short* Bs = &SMEM[p * TILE + ATILE];
    if (issuer) {
      if (ki + 1 < nk) {  // DMA next tile into the other buffer
        const int kn = k0 + BK;
        const unsigned nb = (unsigned)((p ^ 1) * TILE) * 2u;
        tdm_load_tile_2d(smem_base + nb, A + (long)bm * lda + kn, BK, BM,
                         (unsigned long long)lda, (unsigned)(K - kn),
                         (unsigned)(M - bm));
        tdm_load_tile_2d(smem_base + nb + 2u * ATILE, Bt + (long)bn * K + kn,
                         BK, BN, (unsigned long long)K, (unsigned)(K - kn),
                         (unsigned)(N > bn ? N - bn : 0));
        // two DMAs of the *next* tile may stay in flight; current tile done:
        __builtin_amdgcn_s_wait_tensorcnt((short)2);
      } else {
        __builtin_amdgcn_s_wait_tensorcnt((short)0);
      }
    }
    __syncthreads();
#else
    unsigned short* As = &SMEM[0];
    unsigned short* Bs = &SMEM[ATILE];
    // --- stage A tile (128x32): 2 x 16B per thread ---
#pragma unroll
    for (int i = 0; i < 2; ++i) {
      int v = tid + i * 256;        // 0..511
      int r = v >> 2;               // row 0..127
      int c = (v & 3) << 3;         // col 0,8,16,24
      *reinterpret_cast<U16B*>(&As[r * BK + c]) =
          *reinterpret_cast<const U16B*>(A + (long)(bm + r) * lda + k0 + c);
    }
    // --- stage Bt tile (64x32): 1 x 16B per thread, zero-fill n>=N ---
    {
      int r = tid >> 2;
      int c = (tid & 3) << 3;
      U16B val{0ull, 0ull};
      if (bn + r < N)
        val = *reinterpret_cast<const U16B*>(Bt + (long)(bn + r) * K + k0 + c);
      *reinterpret_cast<U16B*>(&Bs[r * BK + c]) = val;
    }
    if (k0 + BK < K) {
      int r = tid >> 1;  // global_prefetch_b8 of next A tile
      __builtin_prefetch(A + (long)(bm + r) * lda + k0 + BK, 0, 3);
    }
    __syncthreads();
#endif

    // --- build fragments per documented CDNA5 layouts ---
    bf16x16 afrag[2], bfrag[2];
#pragma unroll
    for (int i = 0; i < 2; ++i) {
      int mrow = wm * 32 + i * 16 + l16;   // A: lane -> row M, halves split K
      bf16x8 lo = *reinterpret_cast<const bf16x8*>(&As[mrow * BK + 8 * half]);
      bf16x8 hi = *reinterpret_cast<const bf16x8*>(&As[mrow * BK + 16 + 8 * half]);
#pragma unroll
      for (int e = 0; e < 8; ++e) { afrag[i][e] = lo[e]; afrag[i][e + 8] = hi[e]; }
    }
#pragma unroll
    for (int j = 0; j < 2; ++j) {
      int ncol = wn * 32 + j * 16 + l16;   // B from N-major LDS: K = 8*half + e, 16 + 8*half + e
      bf16x8 lo = *reinterpret_cast<const bf16x8*>(&Bs[ncol * BK + 8 * half]);
      bf16x8 hi = *reinterpret_cast<const bf16x8*>(&Bs[ncol * BK + 16 + 8 * half]);
#pragma unroll
      for (int e = 0; e < 8; ++e) { bfrag[j][e] = lo[e]; bfrag[j][e + 8] = hi[e]; }
    }
#pragma unroll
    for (int i = 0; i < 2; ++i)
#pragma unroll
      for (int j = 0; j < 2; ++j)
        acc[i][j] = __builtin_amdgcn_wmma_f32_16x16x32_bf16(
            false, afrag[i], false, bfrag[j], (short)0, acc[i][j], false, false);
    __syncthreads();
  }

  // --- store (C/D layout: VGPR r -> M = r + 8*half, N = lane&15) ---
#pragma unroll
  for (int i = 0; i < 2; ++i)
#pragma unroll
    for (int j = 0; j < 2; ++j)
#pragma unroll
      for (int r = 0; r < 8; ++r) {
        int m = bm + wm * 32 + i * 16 + half * 8 + r;
        int n = bn + wn * 32 + j * 16 + l16;
        if (n < N) {
          float v = acc[i][j][r];
          if (EPI == 1) {
            v += bias[n];
            v = (v > 20.0f) ? v : __logf(1.0f + __expf(v));
          }
          if (ST == 0) {
            C[(long)m * ldc + n] = v;
          } else {  // (b,e,l) transposed store; SEQ==2048
            int b = m >> 11, l = m & 2047;
            C[((long)b * ldc + n) * SEQ + l] = v;
          }
        }
      }
}

// ---------------------------------------------------------------------------
// Depthwise causal conv (width 4) + SiLU, writes bf16 in (b,l,d) layout.
// dir=1 operates on the time-reversed sequence.
// ---------------------------------------------------------------------------
__global__ __launch_bounds__(256) void conv_silu_kernel(
    const float* __restrict__ xz,          // (B, 2*DINNER, SEQ), x = ch 0..2047
    const float* __restrict__ w,           // (DINNER,1,4)
    const float* __restrict__ bc,          // (DINNER)
    unsigned short* __restrict__ xcT,      // bf16 (BL, DINNER)
    int dir) {
  long idx = (long)blockIdx.x * 256 + threadIdx.x;  // d fastest
  if (idx >= (long)BL * DINNER) return;
  int d = (int)(idx & (DINNER - 1));
  int l = (int)((idx >> 11) & (SEQ - 1));
  int b = (int)(idx >> 22);
  float acc = bc[d];
#pragma unroll
  for (int k = 0; k < 4; ++k) {
    int j = l - 3 + k;
    if (j >= 0) {
      int lt = dir ? (SEQ - 1 - j) : j;
      acc += w[d * 4 + k] * xz[((long)b * (2 * DINNER) + d) * SEQ + lt];
    }
  }
  xcT[idx] = f32_to_bf16(silu(acc));
}

// ---------------------------------------------------------------------------
// Selective scan. One thread per (b,d); 16-state recurrence in registers.
// B_t / C_t broadcast through LDS per timestep.
// ---------------------------------------------------------------------------
__global__ __launch_bounds__(256) void scan_kernel(
    const float* __restrict__ xdbl,        // (BL, 96): [.,64..79]=B, [.,80..95]=C
    const float* __restrict__ delta,       // (BL, DINNER)
    const unsigned short* __restrict__ xcT,// bf16 (BL, DINNER)
    const float* __restrict__ xz,          // z = channels 2048..4095
    const float* __restrict__ A_log,       // (DINNER, 16)
    const float* __restrict__ Dp,          // (DINNER)
    float* __restrict__ yout,              // (BL, DINNER)
    int dir) {
  __shared__ float sB[DSTATE];
  __shared__ float sC[DSTATE];
  const int b = blockIdx.y;
  const int d = blockIdx.x * 256 + threadIdx.x;

  float Arow[DSTATE];
#pragma unroll
  for (int s = 0; s < DSTATE; ++s) Arow[s] = -__expf(A_log[d * DSTATE + s]);
  const float Dd = Dp[d];
  float h[DSTATE];
#pragma unroll
  for (int s = 0; s < DSTATE; ++s) h[s] = 0.0f;

  const long zbase = ((long)b * (2 * DINNER) + DINNER + d) * SEQ;

  for (int l = 0; l < SEQ; ++l) {
    if (threadIdx.x < 32) {
      int s = threadIdx.x;
      float v = xdbl[((long)b * SEQ + l) * EDIM + DTRANK + s];
      if (s < DSTATE) sB[s] = v; else sC[s - DSTATE] = v;
    }
    __syncthreads();

    const long p  = ((long)b * SEQ + l) * DINNER + d;
    const float dt = delta[p];
    const float xv = bf16_to_f32(xcT[p]);
    const float dx = dt * xv;
    float y = 0.0f;
#pragma unroll
    for (int s = 0; s < DSTATE; ++s) {
      h[s] = __expf(dt * Arow[s]) * h[s] + dx * sB[s];
      y += h[s] * sC[s];
    }
    y += Dd * xv;
    const int lz = dir ? (SEQ - 1 - l) : l;
    y *= silu(xz[zbase + lz]);
    yout[p] = y;
    __syncthreads();
  }
}

// ---------------------------------------------------------------------------
// out = 0.5*(y_f + reverse_time(y_b)), cast to bf16 (b,l,d) for final GEMM.
// ---------------------------------------------------------------------------
__global__ __launch_bounds__(256) void combine_kernel(
    const float* __restrict__ yf, const float* __restrict__ yb,
    unsigned short* __restrict__ comb) {
  long idx = (long)blockIdx.x * 256 + threadIdx.x;
  if (idx >= (long)BL * DINNER) return;
  int d = (int)(idx & (DINNER - 1));
  int l = (int)((idx >> 11) & (SEQ - 1));
  long b = idx >> 22;
  long pb = ((b * SEQ + (SEQ - 1 - l)) << 11) + d;
  comb[idx] = f32_to_bf16(0.5f * (yf[idx] + yb[pb]));
}

// ---------------------------------------------------------------------------
// host side
// ---------------------------------------------------------------------------
static inline long cdiv(long a, long b) { return (a + b - 1) / b; }

extern "C" void kernel_launch(void* const* d_in, const int* in_sizes, int n_in,
                              void* d_out, int out_size, void* d_ws, size_t ws_size,
                              hipStream_t stream) {
  const float* hid      = (const float*)d_in[0];
  const float* inproj_w = (const float*)d_in[1];
  const float* outproj_w= (const float*)d_in[2];
  const float* conv_w[2]  = {(const float*)d_in[3],  (const float*)d_in[10]};
  const float* conv_b[2]  = {(const float*)d_in[4],  (const float*)d_in[11]};
  const float* xproj_w[2] = {(const float*)d_in[5],  (const float*)d_in[12]};
  const float* dt_w[2]    = {(const float*)d_in[6],  (const float*)d_in[13]};
  const float* dt_b[2]    = {(const float*)d_in[7],  (const float*)d_in[14]};
  const float* A_log[2]   = {(const float*)d_in[8],  (const float*)d_in[15]};
  const float* Dp[2]      = {(const float*)d_in[9],  (const float*)d_in[16]};
  float* out = (float*)d_out;

  char* ws = (char*)d_ws;
  auto suballoc = [&](size_t bytes) -> char* {
    char* p = ws;
    ws += (bytes + 255) & ~(size_t)255;
    return p;
  };
  unsigned short* hid_b   = (unsigned short*)suballoc((size_t)BL * DMODEL * 2);
  unsigned short* win_b   = (unsigned short*)suballoc((size_t)2 * DINNER * DMODEL * 2);
  unsigned short* wxp_b[2], *wdt_b[2];
  wxp_b[0] = (unsigned short*)suballoc((size_t)EDIM * DINNER * 2);
  wxp_b[1] = (unsigned short*)suballoc((size_t)EDIM * DINNER * 2);
  wdt_b[0] = (unsigned short*)suballoc((size_t)DINNER * DTRANK * 2);
  wdt_b[1] = (unsigned short*)suballoc((size_t)DINNER * DTRANK * 2);
  unsigned short* wout_b  = (unsigned short*)suballoc((size_t)DMODEL * DINNER * 2);
  float* xz    = (float*)suballoc((size_t)BATCH * 2 * DINNER * SEQ * 4);
  unsigned short* xcT = (unsigned short*)suballoc((size_t)BL * DINNER * 2);
  float* xdbl  = (float*)suballoc((size_t)BL * EDIM * 4);
  unsigned short* dtr = (unsigned short*)suballoc((size_t)BL * DTRANK * 2);
  float* delta = (float*)suballoc((size_t)BL * DINNER * 4);
  float* yf    = (float*)suballoc((size_t)BL * DINNER * 4);
  float* yb    = (float*)suballoc((size_t)BL * DINNER * 4);
  unsigned short* comb = (unsigned short*)suballoc((size_t)BL * DINNER * 2);

  auto cast = [&](const float* s, unsigned short* d, long n) {
    cast_f2b<<<dim3((unsigned)cdiv(n, 256)), 256, 0, stream>>>(s, d, n);
  };
  cast(hid,       hid_b,  (long)BL * DMODEL);
  cast(inproj_w,  win_b,  (long)2 * DINNER * DMODEL);
  cast(outproj_w, wout_b, (long)DMODEL * DINNER);
  for (int dir = 0; dir < 2; ++dir) {
    cast(xproj_w[dir], wxp_b[dir], (long)EDIM * DINNER);
    cast(dt_w[dir],    wdt_b[dir], (long)DINNER * DTRANK);
  }

  // in_proj: (BL x 1024) x (4096 x 1024)^T -> xz stored (b, e, l)
  wmma_gemm<0, 1><<<dim3(2 * DINNER / 64, BL / 128), 256, 0, stream>>>(
      hid_b, DMODEL, win_b, xz, 2 * DINNER, BL, 2 * DINNER, DMODEL, nullptr);

  const long nEl = (long)BL * DINNER;
  for (int dir = 0; dir < 2; ++dir) {
    conv_silu_kernel<<<dim3((unsigned)cdiv(nEl, 256)), 256, 0, stream>>>(
        xz, conv_w[dir], conv_b[dir], xcT, dir);

    // x_dbl: (BL x 2048) x (96 x 2048)^T -> (BL x 96)
    wmma_gemm<0, 0><<<dim3(cdiv(EDIM, 64), BL / 128), 256, 0, stream>>>(
        xcT, DINNER, wxp_b[dir], xdbl, EDIM, BL, EDIM, DINNER, nullptr);

    // dt_r (bf16 gather of first 64 cols)
    cast_cols<<<dim3((unsigned)cdiv((long)BL * DTRANK, 256)), 256, 0, stream>>>(
        xdbl, dtr, BL, EDIM, DTRANK);

    // delta = softplus((BL x 64) x (2048 x 64)^T + dt_b)
    wmma_gemm<1, 0><<<dim3(DINNER / 64, BL / 128), 256, 0, stream>>>(
        dtr, DTRANK, wdt_b[dir], delta, DINNER, BL, DINNER, DTRANK, dt_b[dir]);

    scan_kernel<<<dim3(DINNER / 256, BATCH), 256, 0, stream>>>(
        xdbl, delta, xcT, xz, A_log[dir], Dp[dir], dir ? yb : yf, dir);
  }

  combine_kernel<<<dim3((unsigned)cdiv(nEl, 256)), 256, 0, stream>>>(yf, yb, comb);

  // out: (BL x 2048) x (1024 x 2048)^T -> (B, L, 1024)
  wmma_gemm<0, 0><<<dim3(DMODEL / 64, BL / 128), 256, 0, stream>>>(
      comb, DINNER, wout_b, out, DMODEL, BL, DMODEL, DINNER, nullptr);
}
